// SwinEncoder3D_70385924047150
// MI455X (gfx1250) — matrix operand order repaired and checked by reading, not compiled, not run on hardware
//
#include <hip/hip_runtime.h>
#include <math.h>

// ---------------------------------------------------------------------------
// Types for CDNA5 WMMA (wave32, 16x16x32 bf16 -> f32)
// ---------------------------------------------------------------------------
typedef __attribute__((ext_vector_type(16))) __bf16        bf16x16;
typedef __attribute__((ext_vector_type(8)))  float         f32x8;
typedef __attribute__((ext_vector_type(4)))  float         f32x4;
typedef __attribute__((ext_vector_type(4)))  unsigned int  uint4v;

union FragB {
  unsigned int u[8];
  uint4v       q[2];
  bf16x16      v;
};

#define DEV __device__ __forceinline__

DEV unsigned short f2bf(float f) {
  return __builtin_bit_cast(unsigned short, (__bf16)f);
}
DEV unsigned int pack2bf(float a, float b) {
  return (unsigned int)f2bf(a) | ((unsigned int)f2bf(b) << 16);
}

// region class along one axis for the shifted-window mask (shift = 3, ws = 7)
DEV int mask_cls(int c, int d) { return (c < d - 7) ? 0 : ((c < d - 3) ? 1 : 2); }

// ---------------------------------------------------------------------------
// Patch embed: Conv3d(4->48, k=4, s=4) NCDHW -> channel-last [B,28,28,28,48]
// ---------------------------------------------------------------------------
__global__ void patch_embed_k(const float* __restrict__ x, const float* __restrict__ w,
                              const float* __restrict__ bias, float* __restrict__ cur) {
  long t = (long)blockIdx.x * blockDim.x + threadIdx.x;
  const long T = 2L * 28 * 28 * 28 * 48;
  if (t >= T) return;
  int oc = (int)(t % 48); long s = t / 48;
  int xw = (int)(s % 28); s /= 28;
  int yw = (int)(s % 28); s /= 28;
  int zw = (int)(s % 28); int b = (int)(s / 28);
  float acc = bias[oc];
  for (int ic = 0; ic < 4; ic++)
    for (int kz = 0; kz < 4; kz++)
      for (int ky = 0; ky < 4; ky++) {
        const float* xr = &x[(((long)(b * 4 + ic) * 112 + (zw * 4 + kz)) * 112 +
                              (yw * 4 + ky)) * 112 + xw * 4];
        const float* wr = &w[(((oc * 4 + ic) * 4 + kz) * 4 + ky) * 4];
        f32x4 xv = *(const f32x4*)xr;
        acc += xv[0] * wr[0] + xv[1] * wr[1] + xv[2] * wr[2] + xv[3] * wr[3];
      }
  cur[t] = acc;
}

// ---------------------------------------------------------------------------
// proj_out: channel LayerNorm (no affine) of channel-last cur -> NCDHW f32
// ---------------------------------------------------------------------------
__global__ void proj_out_k(const float* __restrict__ cur, float* __restrict__ out,
                           int Bn, int D, int H, int Wd, int C) {
  int t = blockIdx.x * blockDim.x + threadIdx.x;
  int T = Bn * D * H * Wd;
  if (t >= T) return;
  const float* row = cur + (long)t * C;
  f32x4 s4 = {0.f, 0.f, 0.f, 0.f};
  for (int c = 0; c < C; c += 4) s4 += ((const f32x4*)row)[c >> 2];
  float m = (s4[0] + s4[1] + s4[2] + s4[3]) / C;
  f32x4 v4 = {0.f, 0.f, 0.f, 0.f};
  for (int c = 0; c < C; c += 4) {
    f32x4 d = ((const f32x4*)row)[c >> 2] - m;
    v4 += d * d;
  }
  float v = (v4[0] + v4[1] + v4[2] + v4[3]) / C;
  float inv = rsqrtf(v + 1e-5f);
  int xx = t % Wd; int s = t / Wd;
  int yy = s % H;  s /= H;
  int zz = s % D;  int b = s / D;
  for (int c = 0; c < C; c++)
    out[(((long)(b * C + c) * D + zz) * H + yy) * Wd + xx] = (row[c] - m) * inv;
}

// ---------------------------------------------------------------------------
// LayerNorm(affine) + optional shift-roll + window-partition. Output bf16 rows.
// windowed=0 -> identity token order (for the MLP path).
// ---------------------------------------------------------------------------
__global__ void ln_part_k(const float* __restrict__ x, const float* __restrict__ g,
                          const float* __restrict__ be, unsigned short* __restrict__ out,
                          int Bn, int D, int H, int Wd, int C, int shift, int windowed) {
  int t = blockIdx.x * blockDim.x + threadIdx.x;
  int T = Bn * D * H * Wd;
  if (t >= T) return;
  int srcRow;
  if (windowed) {
    int n  = t % 343;
    int b_ = t / 343;
    int nWx = Wd / 7, nWy = H / 7, nWz = D / 7;
    int wx = b_ % nWx; int tmp = b_ / nWx;
    int wy = tmp % nWy; tmp /= nWy;
    int wz = tmp % nWz; int b = tmp / nWz;
    int tx = n % 7, ty = (n / 7) % 7, tz = n / 49;
    int z = wz * 7 + tz, y = wy * 7 + ty, xx = wx * 7 + tx;
    int zs = (z + shift) % D, ys = (y + shift) % H, xs = (xx + shift) % Wd;
    srcRow = ((b * D + zs) * H + ys) * Wd + xs;
  } else {
    srcRow = t;
  }
  const float* row = x + (long)srcRow * C;
  f32x4 s4 = {0.f, 0.f, 0.f, 0.f};
  for (int c = 0; c < C; c += 4) s4 += ((const f32x4*)row)[c >> 2];
  float m = (s4[0] + s4[1] + s4[2] + s4[3]) / C;
  f32x4 v4 = {0.f, 0.f, 0.f, 0.f};
  for (int c = 0; c < C; c += 4) {
    f32x4 d = ((const f32x4*)row)[c >> 2] - m;
    v4 += d * d;
  }
  float v = (v4[0] + v4[1] + v4[2] + v4[3]) / C;
  float inv = rsqrtf(v + 1e-5f);
  unsigned int* o32 = (unsigned int*)(out + (long)t * C);
  for (int c = 0; c < C; c += 4) {
    f32x4 xv = ((const f32x4*)row)[c >> 2];
    f32x4 gg = ((const f32x4*)g)[c >> 2];
    f32x4 bb = ((const f32x4*)be)[c >> 2];
    f32x4 yv = (xv - m) * inv * gg + bb;
    o32[(c >> 1)]     = pack2bf(yv[0], yv[1]);
    o32[(c >> 1) + 1] = pack2bf(yv[2], yv[3]);
  }
}

// ---------------------------------------------------------------------------
// Window-reverse + un-roll + residual add: cur[b,z,y,x,c] += yw[windowed]
// ---------------------------------------------------------------------------
__global__ void win_scatter_add_k(const float* __restrict__ yw, float* __restrict__ cur,
                                  int Bn, int D, int H, int Wd, int C, int shift) {
  long t = (long)blockIdx.x * blockDim.x + threadIdx.x;
  long T = (long)Bn * D * H * Wd * C;
  if (t >= T) return;
  int c = (int)(t % C); long wt = t / C;
  int n  = (int)(wt % 343);
  int b_ = (int)(wt / 343);
  int nWx = Wd / 7, nWy = H / 7, nWz = D / 7;
  int wx = b_ % nWx; int tmp = b_ / nWx;
  int wy = tmp % nWy; tmp /= nWy;
  int wz = tmp % nWz; int b = tmp / nWz;
  int tx = n % 7, ty = (n / 7) % 7, tz = n / 49;
  int z = wz * 7 + tz, y = wy * 7 + ty, xx = wx * 7 + tx;
  int zd = (z + shift) % D, yd = (y + shift) % H, xd = (xx + shift) % Wd;
  cur[((long)((b * D + zd) * H + yd) * Wd + xd) * C + c] += yw[t];
}

// ---------------------------------------------------------------------------
// Generic WMMA GEMM: out[M,N] = A[M,K](bf16) * W[N,K](f32)^T (+bias)(+gelu)(+resid)
// 4 waves per block, 64x64 tile, LDS staged, 16x16x32 bf16 WMMA.
// REQUIRES: K % 16 == 0 (holds for all call sites: 48..1536).
// A tile is staged with CDNA5 async global->LDS DMA (ASYNCcnt).
// ---------------------------------------------------------------------------
__global__ __launch_bounds__(128)
void gemm_bf16_wmma(const unsigned short* __restrict__ A, const float* __restrict__ W,
                    const float* __restrict__ bias, const float* __restrict__ resid,
                    float* __restrict__ outF, unsigned short* __restrict__ outB,
                    int M, int N, int K, int act) {
  __shared__ unsigned short lA[64][32];   // [m][k] bf16, 64B rows
  __shared__ unsigned short lB[64][32];   // [n][k] bf16 (W is already [N,K])
  const int m0 = blockIdx.x * 64;
  const int n0 = blockIdx.y * 64;
  const int lane = threadIdx.x & 31;
  const int wave = threadIdx.x >> 5;
  const int wm = (wave >> 1) * 32;
  const int wn = (wave & 1) * 32;
  const int r  = lane & 15;
  const int hi = lane >> 4;
  const int fr = threadIdx.x >> 1;            // fill row 0..63
  const int fs = (threadIdx.x & 1) * 16;      // fill segment (16 elems)

  const f32x8 zero = {0.f, 0.f, 0.f, 0.f, 0.f, 0.f, 0.f, 0.f};
  const uint4v zu  = {0u, 0u, 0u, 0u};
  f32x8 acc[2][2];
  for (int i = 0; i < 2; i++)
    for (int j = 0; j < 2; j++) acc[i][j] = zero;

  const int gr = m0 + fr;
  const int gn = n0 + fr;
  const unsigned int ldsA = (unsigned int)(size_t)&lA[fr][fs];

  for (int k0 = 0; k0 < K; k0 += 32) {
    const bool kin = (k0 + fs) < K;           // K%16==0 -> segment all-in or all-out
    // ---- A tile: async DMA global -> LDS (32 bytes per thread) ----
    if (gr < M && kin) {
      unsigned long ga = (unsigned long)(size_t)&A[(long)gr * K + k0 + fs];
      if (k0 + 32 < K) __builtin_prefetch(&A[(long)gr * K + k0 + fs + 32], 0, 1);
      asm volatile("global_load_async_to_lds_b128 %0, %1, off"
                   :: "v"(ldsA), "v"(ga) : "memory");
      asm volatile("global_load_async_to_lds_b128 %0, %1, off offset:16"
                   :: "v"(ldsA), "v"(ga) : "memory");
    } else {
      ((uint4v*)&lA[fr][fs])[0] = zu;
      ((uint4v*)&lA[fr][fs])[1] = zu;
    }
    // ---- W tile: vector load f32, convert, ds_store_b128 ----
    if (gn < N && kin) {
      const f32x4* wp = (const f32x4*)&W[(long)gn * K + k0 + fs];
      f32x4 w0 = wp[0], w1 = wp[1], w2 = wp[2], w3 = wp[3];
      uint4v p0, p1;
      p0[0] = pack2bf(w0[0], w0[1]); p0[1] = pack2bf(w0[2], w0[3]);
      p0[2] = pack2bf(w1[0], w1[1]); p0[3] = pack2bf(w1[2], w1[3]);
      p1[0] = pack2bf(w2[0], w2[1]); p1[1] = pack2bf(w2[2], w2[3]);
      p1[2] = pack2bf(w3[0], w3[1]); p1[3] = pack2bf(w3[2], w3[3]);
      ((uint4v*)&lB[fr][fs])[0] = p0;
      ((uint4v*)&lB[fr][fs])[1] = p1;
    } else {
      ((uint4v*)&lB[fr][fs])[0] = zu;
      ((uint4v*)&lB[fr][fs])[1] = zu;
    }
    asm volatile("s_wait_asynccnt 0x0" ::: "memory");
    __syncthreads();

    FragB a[2], b[2];
    for (int i = 0; i < 2; i++) {
      const uint4v* rowA = (const uint4v*)&lA[wm + i * 16 + r][0];
      a[i].q[0] = rowA[hi];          // K {0..7} / {8..15}
      a[i].q[1] = rowA[hi + 2];      // K {16..23} / {24..31}
      const uint4v* rowB = (const uint4v*)&lB[wn + i * 16 + r][0];
      b[i].q[0] = rowB[hi];          // K {0..7} / {8..15}
      b[i].q[1] = rowB[hi + 2];      // K {16..23} / {24..31}
    }
    for (int i = 0; i < 2; i++)
      for (int j = 0; j < 2; j++)
        acc[i][j] = __builtin_amdgcn_wmma_f32_16x16x32_bf16(
            false, a[i].v, false, b[j].v, (short)0, acc[i][j], false, false);
    __syncthreads();
  }

  for (int i = 0; i < 2; i++) {
    for (int j = 0; j < 2; j++) {
      int col = n0 + wn + j * 16 + r;
      if (col >= N) continue;
      float bv = bias ? bias[col] : 0.f;
      for (int v = 0; v < 8; v++) {
        int row = m0 + wm + i * 16 + hi * 8 + v;
        if (row >= M) continue;
        float xv = acc[i][j][v] + bv;
        if (act == 1) xv = 0.5f * xv * (1.f + erff(xv * 0.70710678118654752f));
        long o = (long)row * N + col;
        if (resid) xv += resid[o];
        if (outB) outB[o] = f2bf(xv);
        else      outF[o] = xv;
      }
    }
  }
}

// ---------------------------------------------------------------------------
// Window attention core. One wave per (q-block, head, window).
// qkv bf16 [B_,343,3,Hh,16]; out bf16 [B_,343,C]. hd = 16 (padded to K=32).
// Relative-position bias and shift mask computed analytically.
// ---------------------------------------------------------------------------
__global__ __launch_bounds__(32)
void win_attn_k(const unsigned short* __restrict__ qkv, const float* __restrict__ rpb,
                unsigned short* __restrict__ out, int C, int Hh,
                int D, int H, int Wd, int shift) {
  __shared__ float          S[16][352];
  __shared__ unsigned short P[16][352];
  const int qb   = blockIdx.x;   // 0..21
  const int h    = blockIdx.y;
  const int b_   = blockIdx.z;
  const int lane = threadIdx.x;
  const int r    = lane & 15;
  const int hi   = lane >> 4;
  const int NN   = 343;
  const f32x8 zero = {0.f, 0.f, 0.f, 0.f, 0.f, 0.f, 0.f, 0.f};

  int nWx = Wd / 7, nWy = H / 7;
  int wx = b_ % nWx; int tq = b_ / nWx;
  int wy = tq % nWy; tq /= nWy;
  int wz = tq % (D / 7);

  // ---- Q fragment (A matrix, 16 rows x K32 with dd 16..31 zero) ----
  FragB qa;
  for (int i = 0; i < 8; i++) qa.u[i] = 0;
  int qi = qb * 16 + r;
  if (qi < NN) {
    const uint4v* qp = (const uint4v*)(qkv + (((long)b_ * NN + qi) * 3 + 0) * C + h * 16);
    qa.q[0] = qp[hi];
  }

  // ---- S = scale*Q.K^T + bias + mask ----
  for (int jb = 0; jb < 22; jb++) {
    FragB kb;
    for (int i = 0; i < 8; i++) kb.u[i] = 0;
    {
      int j = jb * 16 + r;
      if (j < NN) {
        const uint4v* kp = (const uint4v*)(qkv + (((long)b_ * NN + j) * 3 + 1) * C + h * 16);
        kb.q[0] = kp[hi];
      }
    }
    f32x8 sa = zero;
    sa = __builtin_amdgcn_wmma_f32_16x16x32_bf16(false, qa.v, false, kb.v, (short)0, sa,
                                                 false, false);
    int j = jb * 16 + r;
    int jz = j / 49, jy = (j / 7) % 7, jx = j % 7;
    for (int v = 0; v < 8; v++) {
      int il = v + hi * 8;
      int i  = qb * 16 + il;
      float val = -1e9f;
      if (j < NN) {
        if (i < NN) {
          int iz = i / 49, iy = (i / 7) % 7, ix = i % 7;
          int rp = (((iz - jz + 6) * 13 + (iy - jy + 6)) * 13 + (ix - jx + 6));
          float bias = rpb[(long)rp * Hh + h];
          float mk = 0.f;
          if (shift) {
            int ri = (mask_cls(wz * 7 + iz, D) * 3 + mask_cls(wy * 7 + iy, H)) * 3 +
                     mask_cls(wx * 7 + ix, Wd);
            int rj = (mask_cls(wz * 7 + jz, D) * 3 + mask_cls(wy * 7 + jy, H)) * 3 +
                     mask_cls(wx * 7 + jx, Wd);
            if (ri != rj) mk = -100.f;
          }
          val = sa[v] * 0.25f + bias + mk;
        } else {
          val = 0.f;  // unused row, keep finite
        }
      }
      S[il][jb * 16 + r] = val;
    }
  }
  __syncthreads();

  // ---- row softmax -> P (bf16); lanes (r, r+16) split each row ----
  {
    int j0 = hi * 176;
    float mx = -1e30f;
    for (int j = 0; j < 176; j++) mx = fmaxf(mx, S[r][j0 + j]);
    mx = fmaxf(mx, __shfl_xor(mx, 16, 32));
    float sum = 0.f;
    for (int j = 0; j < 176; j++) {
      float e = __expf(S[r][j0 + j] - mx);
      S[r][j0 + j] = e;
      sum += e;
    }
    sum += __shfl_xor(sum, 16, 32);
    float inv = 1.f / sum;
    for (int j = 0; j < 176; j += 2)
      ((unsigned int*)&P[r][0])[(j0 + j) >> 1] =
          pack2bf(S[r][j0 + j] * inv, S[r][j0 + j + 1] * inv);
  }
  __syncthreads();

  // ---- out = P @ V ----
  f32x8 oacc = zero;
  for (int kk = 0; kk < 11; kk++) {
    FragB pa;
    const uint4v* pr = (const uint4v*)&P[r][kk * 32];
    pa.q[0] = pr[2 * hi];
    pa.q[1] = pr[2 * hi + 1];
    FragB vb;
    const unsigned short* vbase = qkv + ((long)b_ * NN * 3 + 2) * C + h * 16 + r;
    const long stride3C = 3L * C;
    if (kk < 10) {  // tokens <= 319 < 343: no guards needed
      for (int v = 0; v < 8; v++) {
        int t0 = kk * 32 + hi * 16 + 2 * v;
        unsigned int lo   = vbase[(long)t0 * stride3C];
        unsigned int hi16 = vbase[(long)(t0 + 1) * stride3C];
        vb.u[v] = lo | (hi16 << 16);
      }
    } else {
      for (int v = 0; v < 8; v++) {
        int t0 = kk * 32 + hi * 16 + 2 * v;
        unsigned int lo = 0, hi16 = 0;
        if (t0 < NN)     lo   = vbase[(long)t0 * stride3C];
        if (t0 + 1 < NN) hi16 = vbase[(long)(t0 + 1) * stride3C];
        vb.u[v] = lo | (hi16 << 16);
      }
    }
    oacc = __builtin_amdgcn_wmma_f32_16x16x32_bf16(false, pa.v, false, vb.v, (short)0, oacc,
                                                   false, false);
  }
  for (int v = 0; v < 8; v++) {
    int i = qb * 16 + v + hi * 8;
    if (i < NN) out[((long)b_ * NN + i) * C + h * 16 + r] = f2bf(oacc[v]);
  }
}

// ---------------------------------------------------------------------------
// Patch merging: gather 8 shifted slices (with MONAI's duplicate quirk, zero
// padding for odd dims) + LayerNorm(affine) -> bf16 [T2, 8C]
// ---------------------------------------------------------------------------
__global__ void merge_gather_ln_k(const float* __restrict__ cur, const float* __restrict__ g,
                                  const float* __restrict__ be, unsigned short* __restrict__ out,
                                  int Bn, int D, int H, int Wd, int C) {
  int D2 = (D + 1) / 2, H2 = (H + 1) / 2, W2 = (Wd + 1) / 2;
  int t = blockIdx.x * blockDim.x + threadIdx.x;
  int T = Bn * D2 * H2 * W2;
  if (t >= T) return;
  int x2 = t % W2; int s = t / W2;
  int y2 = s % H2; s /= H2;
  int z2 = s % D2; int b = s / D2;
  const int offs[8][3] = {{0,0,0},{1,0,0},{0,1,0},{0,0,1},{1,0,1},{0,1,0},{0,0,1},{1,1,1}};
  const int C8 = 8 * C;
  const float* rows[8];
  for (int p = 0; p < 8; p++) {
    int z = 2 * z2 + offs[p][0], y = 2 * y2 + offs[p][1], xx = 2 * x2 + offs[p][2];
    rows[p] = (z < D && y < H && xx < Wd)
                  ? cur + ((long)((b * D + z) * H + y) * Wd + xx) * C
                  : (const float*)0;
  }
  f32x4 s4 = {0.f, 0.f, 0.f, 0.f};
  for (int p = 0; p < 8; p++)
    if (rows[p])
      for (int c = 0; c < C; c += 4) s4 += ((const f32x4*)rows[p])[c >> 2];
  float m = (s4[0] + s4[1] + s4[2] + s4[3]) / C8;
  f32x4 v4 = {0.f, 0.f, 0.f, 0.f};
  float zc = 0.f;  // count of zero-padded groups contributes (0-m)^2
  for (int p = 0; p < 8; p++) {
    if (rows[p]) {
      for (int c = 0; c < C; c += 4) {
        f32x4 d = ((const f32x4*)rows[p])[c >> 2] - m;
        v4 += d * d;
      }
    } else {
      zc += (float)C;
    }
  }
  float v = (v4[0] + v4[1] + v4[2] + v4[3] + zc * m * m) / C8;
  float inv = rsqrtf(v + 1e-5f);
  unsigned int* o32 = (unsigned int*)(out + (long)t * C8);
  for (int p = 0; p < 8; p++) {
    for (int c = 0; c < C; c += 2) {
      float a0 = rows[p] ? rows[p][c]     : 0.f;
      float a1 = rows[p] ? rows[p][c + 1] : 0.f;
      int cc = p * C + c;
      a0 = (a0 - m) * inv * g[cc]     + be[cc];
      a1 = (a1 - m) * inv * g[cc + 1] + be[cc + 1];
      o32[cc >> 1] = pack2bf(a0, a1);
    }
  }
}

// ---------------------------------------------------------------------------
// Host orchestration
// ---------------------------------------------------------------------------
struct BlkP { const float *fc1_b,*fc1_w,*fc2_b,*fc2_w,*n1b,*n1g,*n2b,*n2g,*pb,*pw,*qb,*qw,*rpb; };
struct StgP { BlkP blk[6]; const float *mnb,*mng,*mrw; int depth; };

extern "C" void kernel_launch(void* const* d_in, const int* in_sizes, int n_in,
                              void* d_out, int out_size, void* d_ws, size_t ws_size,
                              hipStream_t stream) {
  (void)in_sizes; (void)n_in; (void)out_size; (void)ws_size;
  // --- unpack params (JAX pytree order: alphabetical dict keys; "params" < "x")
  int idx = 0;
  const float* patch_b = (const float*)d_in[idx++];
  const float* patch_w = (const float*)d_in[idx++];
  StgP st[3];
  const int depths[3] = {2, 2, 6};
  for (int s = 0; s < 3; s++) {
    st[s].depth = depths[s];
    for (int b = 0; b < depths[s]; b++) {
      BlkP& k = st[s].blk[b];
      k.fc1_b = (const float*)d_in[idx++];
      k.fc1_w = (const float*)d_in[idx++];
      k.fc2_b = (const float*)d_in[idx++];
      k.fc2_w = (const float*)d_in[idx++];
      k.n1b   = (const float*)d_in[idx++];
      k.n1g   = (const float*)d_in[idx++];
      k.n2b   = (const float*)d_in[idx++];
      k.n2g   = (const float*)d_in[idx++];
      k.pb    = (const float*)d_in[idx++];
      k.pw    = (const float*)d_in[idx++];
      k.qb    = (const float*)d_in[idx++];
      k.qw    = (const float*)d_in[idx++];
      k.rpb   = (const float*)d_in[idx++];
    }
    st[s].mnb = (const float*)d_in[idx++];
    st[s].mng = (const float*)d_in[idx++];
    st[s].mrw = (const float*)d_in[idx++];
  }
  const float* x = (const float*)d_in[idx++];

  // --- workspace carve-out (regions aliased across phases)
  char* ws = (char*)d_ws;
  size_t off = 0;
  auto alloc = [&](size_t bytes) -> void* {
    void* p = ws + off;
    off += (bytes + 255) & ~(size_t)255;
    return p;
  };
  float*          cur    = (float*)alloc(16u << 20);          // activations [M,C] f32
  unsigned short* lnbuf  = (unsigned short*)alloc(8u << 20);  // ln out / attn out (aliased)
  unsigned short* qkvbuf = (unsigned short*)alloc(24u << 20); // qkv / mlp hidden (aliased)
  float*          ywbuf  = (float*)alloc(16u << 20);          // windowed proj out / merge-in

  unsigned short* attnbuf  = lnbuf;                 // alias: ln dead after qkv gemm
  unsigned short* hidden   = qkvbuf;                // alias: qkv dead after attention
  unsigned short* mergein  = (unsigned short*)ywbuf;// alias: yw dead after scatter

  float* outp = (float*)d_out;
  const int heads[3] = {3, 6, 12};

  // --- patch embed + feat0
  {
    long T = 2L * 28 * 28 * 28 * 48;
    patch_embed_k<<<(unsigned)((T + 255) / 256), 256, 0, stream>>>(x, patch_w, patch_b, cur);
    int Tt = 2 * 28 * 28 * 28;
    proj_out_k<<<(Tt + 127) / 128, 128, 0, stream>>>(cur, outp, 2, 28, 28, 28, 48);
    outp += (long)Tt * 48;
  }

  int D = 28, C = 48;
  for (int s = 0; s < 3; s++) {
    int Hh = heads[s];
    int nW1 = D / 7;
    int B_  = 2 * nW1 * nW1 * nW1;
    int M   = B_ * 343;                 // == 2*D*D*D
    int shiftAmt = (D > 7) ? 3 : 0;

    for (int b = 0; b < st[s].depth; b++) {
      BlkP& k = st[s].blk[b];
      int sh = (b & 1) ? shiftAmt : 0;
      // attention branch
      ln_part_k<<<(M + 127) / 128, 128, 0, stream>>>(cur, k.n1g, k.n1b, lnbuf,
                                                     2, D, D, D, C, sh, 1);
      dim3 gq((M + 63) / 64, (3 * C + 63) / 64);
      gemm_bf16_wmma<<<gq, 128, 0, stream>>>(lnbuf, k.qw, k.qb, nullptr,
                                             nullptr, qkvbuf, M, 3 * C, C, 0);
      dim3 ga(22, Hh, B_);
      win_attn_k<<<ga, 32, 0, stream>>>(qkvbuf, k.rpb, attnbuf, C, Hh, D, D, D, sh);
      dim3 gp((M + 63) / 64, (C + 63) / 64);
      gemm_bf16_wmma<<<gp, 128, 0, stream>>>(attnbuf, k.pw, k.pb, nullptr,
                                             ywbuf, nullptr, M, C, C, 0);
      long E = (long)M * C;
      win_scatter_add_k<<<(unsigned)((E + 255) / 256), 256, 0, stream>>>(ywbuf, cur,
                                                                         2, D, D, D, C, sh);
      // MLP branch (token order = spatial, pointwise, residual in place)
      ln_part_k<<<(M + 127) / 128, 128, 0, stream>>>(cur, k.n2g, k.n2b, lnbuf,
                                                     2, D, D, D, C, 0, 0);
      dim3 g1((M + 63) / 64, (4 * C + 63) / 64);
      gemm_bf16_wmma<<<g1, 128, 0, stream>>>(lnbuf, k.fc1_w, k.fc1_b, nullptr,
                                             nullptr, hidden, M, 4 * C, C, 1);
      dim3 g2((M + 63) / 64, (C + 63) / 64);
      gemm_bf16_wmma<<<g2, 128, 0, stream>>>(hidden, k.fc2_w, k.fc2_b, cur,
                                             cur, nullptr, M, C, 4 * C, 0);
    }

    // patch merging
    int D2 = (D + 1) / 2;
    int T2 = 2 * D2 * D2 * D2;
    merge_gather_ln_k<<<(T2 + 127) / 128, 128, 0, stream>>>(cur, st[s].mng, st[s].mnb,
                                                            mergein, 2, D, D, D, C);
    dim3 gm((T2 + 63) / 64, (2 * C + 63) / 64);
    gemm_bf16_wmma<<<gm, 128, 0, stream>>>(mergein, st[s].mrw, nullptr, nullptr,
                                           cur, nullptr, T2, 2 * C, 8 * C, 0);
    D = D2;
    C = 2 * C;
    proj_out_k<<<(T2 + 127) / 128, 128, 0, stream>>>(cur, outp, 2, D, D, D, C);
    outp += (long)T2 * C;
  }
}
